// EfficientZeroBlock_41437844472392
// MI455X (gfx1250) — matrix operand order repaired and checked by reading, not compiled, not run on hardware
//
#include <hip/hip_runtime.h>

typedef _Float16 half8 __attribute__((ext_vector_type(8)));
typedef _Float16 v16h  __attribute__((ext_vector_type(16)));
typedef float    v8f   __attribute__((ext_vector_type(8)));

#define B_   128
#define C_   64
#define T_   512
#define V_   25
#define K_   3
#define TT   16                 // t-values per workgroup tile
#define NB   (TT*V_)            // 400 = stage-1 N dimension
#define NTHREADS 256            // 8 wave32
#define NWAVES 8
#define TBLK (T_/TT)            // 32 t-blocks

// ---- dynamic LDS partition (bytes) ----
#define OFF_AT   0                          // A_norm^T f16 [K][32][32] (v,w padded, zero pad)
#define OFF_W    (OFF_AT + K_*32*32*2)      // W f16 [K][64][64]
#define OFF_XT   (OFF_W  + K_*C_*C_*2)      // x^T f16 [NB][64]  ((t,v)-major, c contiguous)
#define OFF_Z    (OFF_XT + NB*C_*2)         // z f16 [1024][32]  ((o*16+t)-major, v padded to 32)
#define OFF_D    (OFF_Z  + (C_*TT)*32*2)    // degree rsqrt f32 [80]
#define OFF_BNS  (OFF_D  + 80*4)            // BN scale f32 [64]
#define OFF_BNB  (OFF_BNS + C_*4)           // BN bias  f32 [64]
#define LDS_BYTES (OFF_BNB + C_*4)          // ~148 KB

__device__ __forceinline__ float hswish(float v) {
  return v * fminf(fmaxf(v + 3.0f, 0.0f), 6.0f) * (1.0f / 6.0f);
}

// A-operand (16x32 f16, M x K): lane holds row (row0 + lane%16);
// halves 0..15 = K (q..q+15), q = (lane<16 ? 0 : 16).
__device__ __forceinline__ v16h load_frag_a(const _Float16* base, int strideH,
                                            int row0, int kbase, int lane) {
  const _Float16* p = base + (row0 + (lane & 15)) * strideH + kbase + ((lane < 16) ? 0 : 16);
  return *(const v16h*)p;
}

// B-operand (32x16 f16, K x N) gathered from transposed storage (N-major rows of K):
// lane holds column (n0 + lane%16); halves = K rows (s..s+15), s = (lane<16 ? 0 : 16).
// 16 contiguous halves -> one 32B (2x ds_load_b128) read.
__device__ __forceinline__ v16h load_frag_b(const _Float16* baseT, int strideH,
                                            int n0, int kbase, int lane) {
  const _Float16* p = baseT + (n0 + (lane & 15)) * strideH + kbase + ((lane < 16) ? 0 : 16);
  return *(const v16h*)p;
}

extern "C" __global__ __launch_bounds__(NTHREADS)
void egcn_fused_kernel(const float* __restrict__ xg,
                       const float* __restrict__ Af,
                       const float* __restrict__ Al,
                       const float* __restrict__ Wg,
                       const float* __restrict__ gamma_,
                       const float* __restrict__ beta_,
                       const float* __restrict__ bmean,
                       const float* __restrict__ bvar,
                       float* __restrict__ outg)
{
  extern __shared__ __align__(32) char smem[];
  _Float16* AT  = (_Float16*)(smem + OFF_AT);
  _Float16* Wl  = (_Float16*)(smem + OFF_W);
  _Float16* xT  = (_Float16*)(smem + OFF_XT);
  _Float16* zl  = (_Float16*)(smem + OFF_Z);
  float*    dL  = (float*)(smem + OFF_D);
  float*    bnS = (float*)(smem + OFF_BNS);
  float*    bnB = (float*)(smem + OFF_BNB);

  const int tid  = threadIdx.x;
  const int lane = tid & 31;         // wave32
  const int wave = tid >> 5;
  const int b    = blockIdx.x >> 5;  // / TBLK
  const int t0   = (blockIdx.x & 31) * TT;

  // --- stage 0a: column-degree d[k][w] = rsqrt(max(sum_v |Af+Al|, eps))
  for (int idx = tid; idx < K_ * V_; idx += NTHREADS) {
    int k = idx / V_, w = idx % V_;
    float s = 0.f;
    for (int v = 0; v < V_; ++v)
      s += fabsf(Af[(k * V_ + v) * V_ + w] + Al[(k * V_ + v) * V_ + w]);
    dL[idx] = rsqrtf(fmaxf(s, 1e-6f));
  }
  // --- stage 0b: BN folded scale/bias
  for (int idx = tid; idx < C_; idx += NTHREADS) {
    float sc = rsqrtf(bvar[idx] + 1e-5f) * gamma_[idx];
    bnS[idx] = sc;
    bnB[idx] = beta_[idx] - bmean[idx] * sc;
  }
  // --- stage 0c: W -> f16 (same (k,o,c) row-major layout)
  for (int idx = tid; idx < K_ * C_ * C_; idx += NTHREADS)
    Wl[idx] = (_Float16)Wg[idx];
  // --- stage 0d: x tile -> f16, transposed to ((t,v), c). Global read is fully coalesced:
  // x[b, c, t0..t0+TT, :] is 400 contiguous floats per channel.
  {
    const float* xb = xg + (((long)(b * C_)) * T_ + t0) * V_;
    for (int idx = tid; idx < C_ * NB; idx += NTHREADS) {
      int c = idx / NB, n = idx % NB;
      xT[n * C_ + c] = (_Float16)xb[(long)c * T_ * V_ + n];
    }
  }
  // --- stage 0e: zero z's v-padding (written once; stage 1 only touches v<25)
  for (int idx = tid; idx < C_ * TT * (32 - V_); idx += NTHREADS) {
    int row = idx / (32 - V_), v = V_ + idx % (32 - V_);
    zl[row * 32 + v] = (_Float16)0.f;
  }
  __syncthreads();
  // --- stage 0f: A_norm^T f16, padded to 32x32, stored [k][w][v] (B^T layout)
  for (int idx = tid; idx < K_ * 32 * 32; idx += NTHREADS) {
    int k = idx >> 10, w = (idx >> 5) & 31, v = idx & 31;
    float val = 0.f;
    if (v < V_ && w < V_) {
      float a = Af[(k * V_ + v) * V_ + w] + Al[(k * V_ + v) * V_ + w];
      val = dL[k * V_ + v] * fabsf(a) * dL[k * V_ + w];
    }
    AT[idx] = (_Float16)val;
  }
  __syncthreads();

  // persistent f32 accumulators: 8 m-tiles x 2 n-tiles per wave (128 VGPRs)
  v8f yacc[16];
#pragma unroll
  for (int i = 0; i < 16; ++i) {
    v8f zz = {0.f, 0.f, 0.f, 0.f, 0.f, 0.f, 0.f, 0.f};
    yacc[i] = zz;
  }

  for (int k = 0; k < K_; ++k) {
    if (k) __syncthreads();  // previous stage-2 reads of z complete before overwrite
    // ---- stage 1: z = W[k] (64x64) @ x (64 x 400); 100 output tiles, K=64 -> 2 WMMA each
    const _Float16* wk = Wl + k * C_ * C_;
    for (int tile = wave; tile < 4 * V_; tile += NWAVES) {
      int otile = tile / V_;
      int ntile = tile % V_;
      v8f acc = {0.f, 0.f, 0.f, 0.f, 0.f, 0.f, 0.f, 0.f};
      v16h a0 = load_frag_a(wk, C_, otile * 16, 0, lane);
      v16h b0 = load_frag_b(xT, C_, ntile * 16, 0, lane);
      acc = __builtin_amdgcn_wmma_f32_16x16x32_f16(false, a0, false, b0, (short)0, acc, false, false);
      v16h a1 = load_frag_a(wk, C_, otile * 16, 32, lane);
      v16h b1 = load_frag_b(xT, C_, ntile * 16, 32, lane);
      acc = __builtin_amdgcn_wmma_f32_16x16x32_f16(false, a1, false, b1, (short)0, acc, false, false);
      // scatter D tile -> z[(o*TT + t), v] as f16 (row = VGPR index + 8*(lane>=16), col = lane%16)
      int ncol = ntile * 16 + (lane & 15);
      int tl = ncol / V_, vv = ncol % V_;
      int radd = (lane < 16) ? 0 : 8;
#pragma unroll
      for (int r = 0; r < 8; ++r) {
        int o = otile * 16 + r + radd;
        zl[(o * TT + tl) * 32 + vv] = (_Float16)acc[r];
      }
    }
    __syncthreads();
    // ---- stage 2: y += z (1024 x 32pad) @ A_norm[k] (32pad x 32pad); 1 WMMA per tile
    v16h bf0 = load_frag_b(AT + k * 1024, 32, 0,  0, lane);
    v16h bf1 = load_frag_b(AT + k * 1024, 32, 16, 0, lane);
#pragma unroll
    for (int j = 0; j < 8; ++j) {
      int mtile = wave * 8 + j;
      v16h af = load_frag_a(zl, 32, mtile * 16, 0, lane);
      yacc[2 * j] =
          __builtin_amdgcn_wmma_f32_16x16x32_f16(false, af, false, bf0, (short)0, yacc[2 * j], false, false);
      yacc[2 * j + 1] =
          __builtin_amdgcn_wmma_f32_16x16x32_f16(false, af, false, bf1, (short)0, yacc[2 * j + 1], false, false);
    }
  }

  // ---- epilogue: y/K -> BN -> hardswish -> +residual -> hardswish -> store
  const float inv3 = 1.0f / (float)K_;
  const int radd = (lane < 16) ? 0 : 8;
#pragma unroll
  for (int j = 0; j < 8; ++j) {
    int mtile = wave * 8 + j;
#pragma unroll
    for (int nt = 0; nt < 2; ++nt) {
      int w = nt * 16 + (lane & 15);
      if (w >= V_) continue;  // padded joint columns
      v8f acc = yacc[2 * j + nt];
#pragma unroll
      for (int r = 0; r < 8; ++r) {
        int m = mtile * 16 + r + radd;  // (o, t_local) row, TT == 16
        int o = m >> 4;
        int tl = m & 15;
        float y = acc[r] * inv3;
        float h = hswish(y * bnS[o] + bnB[o]);
        long gidx = ((long)(b * C_ + o) * T_ + (t0 + tl)) * V_ + w;
        outg[gidx] = hswish(h + xg[gidx]);
      }
    }
  }
}

extern "C" void kernel_launch(void* const* d_in, const int* in_sizes, int n_in,
                              void* d_out, int out_size, void* d_ws, size_t ws_size,
                              hipStream_t stream) {
  (void)in_sizes; (void)n_in; (void)out_size; (void)d_ws; (void)ws_size;
  const float* x  = (const float*)d_in[0];
  const float* Af = (const float*)d_in[1];
  const float* Al = (const float*)d_in[2];
  const float* W  = (const float*)d_in[3];
  const float* g  = (const float*)d_in[4];
  const float* be = (const float*)d_in[5];
  const float* mu = (const float*)d_in[6];
  const float* va = (const float*)d_in[7];
  float* out = (float*)d_out;

  dim3 grid(B_ * TBLK);  // 128 batches x 32 t-blocks = 4096 workgroups
  egcn_fused_kernel<<<grid, NTHREADS, LDS_BYTES, stream>>>(x, Af, Al, W, g, be, mu, va, out);
}
